// SFCIM_87651692577711
// MI455X (gfx1250) — compile-verified, matrix operand order checked by reading, not run on hardware
//
// MI455X / gfx1250 implementation of the SFCIM block.
// GEMMs (in_proj, x_proj, out_proj, lin, FIM MLP, rfft2/irfft2 as DFT GEMMs)
// run on V_WMMA_F32_16X16X32_F16; each wave owns two 16x16 tiles (A-fragment
// reuse). Weight tiles are DMA'd into LDS by the Tensor Data Mover via inline
// asm (tensor_load_to_lds + s_wait_tensorcnt); A tiles are staged with
// vectorized f32 loads + convertvector -> f16 LDS stores. The selective scan
// double-buffers its per-step coefficient row through async global->LDS
// copies (global_load_async_to_lds_b128 + s_wait_asynccnt).
#include <hip/hip_runtime.h>
#include <hip/hip_bf16.h>

#define B_  8
#define N_  4096
#define C_  96
#define H_  64
#define W_  64
#define DI_ 192
#define R_  6
#define NS_ 16
#define K_  4
#define L_  4096

typedef __attribute__((ext_vector_type(16))) _Float16 v16h;
typedef __attribute__((ext_vector_type(8)))  _Float16 v8h;
typedef __attribute__((ext_vector_type(4)))  _Float16 v4h;
typedef __attribute__((ext_vector_type(8)))  float    v8f;
typedef __attribute__((ext_vector_type(4)))  float    v4f;
typedef __attribute__((ext_vector_type(4)))  unsigned int v4u;
typedef __attribute__((ext_vector_type(8)))  unsigned int v8u;

#define TDM_ASM   1   // tensor_load_to_lds via inline asm (B tiles)
#define ASYNC_ASM 1   // global_load_async_to_lds_b128 via inline asm (scan)

// ---------------------------------------------------------------------------
// WMMA GEMM:  O[M,N] (+)= act( A[M,Kd](f32) * Wh[N,Kp](f16)^T + bias )
// 128 threads = 4 waves; block covers 64 rows x 32 cols (2 tiles per wave).
// Requires M%64==0, N%32==0, Kd%32==0, Kd==lda (callers guarantee/pad).
// ---------------------------------------------------------------------------
__launch_bounds__(128)
__global__ void gemm_wmma_f16(const float* __restrict__ A,
                              const _Float16* __restrict__ Wh,
                              const float* __restrict__ bias,
                              float* __restrict__ O,
                              int Kd, int Kp, int ldo, int act, int accmode)
{
    __shared__ __align__(16) _Float16 As[64][32];
    __shared__ __align__(16) _Float16 Bs[32][32];

    const int n0   = blockIdx.x * 32;
    const int row0 = blockIdx.y * 64;
    const int tid  = threadIdx.x;
    const int lane = tid & 31;
    const int wid  = tid >> 5;
    const int hf   = lane >> 4;     // which K-half this lane holds
    const int idx  = lane & 15;     // row (A) / col (B,C)

    const int ar = tid >> 1;              // A staging: row 0..63
    const int aq = (tid & 1) * 16;        // col group 0 / 16

    v8f acc0 = {0.f,0.f,0.f,0.f,0.f,0.f,0.f,0.f};
    v8f acc1 = {0.f,0.f,0.f,0.f,0.f,0.f,0.f,0.f};

    for (int kc = 0; kc < Kd; kc += 32) {
        // ---- stage A tile: 4x float4 load, convert, ds-store f16 ----
        {
            const v4f* pa = (const v4f*)(A + (size_t)(row0 + ar) * Kd + kc + aq);
            v4f f0 = pa[0], f1 = pa[1], f2 = pa[2], f3 = pa[3];
            __builtin_prefetch(pa + 8, 0, 1);    // next K chunk, same row
            *(v4h*)&As[ar][aq +  0] = __builtin_convertvector(f0, v4h);
            *(v4h*)&As[ar][aq +  4] = __builtin_convertvector(f1, v4h);
            *(v4h*)&As[ar][aq +  8] = __builtin_convertvector(f2, v4h);
            *(v4h*)&As[ar][aq + 12] = __builtin_convertvector(f3, v4h);
        }

        // ---- stage W tile (32 rows x 32 f16) via Tensor Data Mover ----
#if TDM_ASM
        if (tid < 32) {   // one wave issues the DMA (TDM ignores EXEC)
            unsigned long long ga =
                (unsigned long long)(const void*)(Wh + (size_t)n0 * Kp + kc);
            v4u g0;
            g0[0] = 1u;                                    // count=1, user mode
            g0[1] = (unsigned int)(unsigned long long)(void*)&Bs[0][0];
            g0[2] = (unsigned int)(ga & 0xffffffffull);
            g0[3] = (unsigned int)((ga >> 32) & 0x1ffffffull) | (2u << 30);
            unsigned int td0 = (unsigned int)Kp;           // tensor_dim0 (elems)
            unsigned int td1 = 32u;                        // tensor_dim1 (rows)
            v8u g1;
            g1[0] = (1u << 16);                            // data_size = 2B
            g1[1] = (td0 & 0xffffu) << 16;                 // tensor_dim0 lo16
            g1[2] = (td0 >> 16) | ((td1 & 0xffffu) << 16);
            g1[3] = (td1 >> 16) | (32u << 16);             // tile_dim0 = 32
            g1[4] = 32u;                                   // tile_dim1 = 32
            g1[5] = td0;                                   // dim0 stride lo
            g1[6] = 0u;
            g1[7] = 0u;
            asm volatile("tensor_load_to_lds %0, %1\n\t"
                         "s_wait_tensorcnt 0x0"
                         :: "s"(g0), "s"(g1) : "memory");
        }
#else
        if (tid < 64) {
            int r = tid >> 1, q = (tid & 1) * 16;
            const v8h* pw = (const v8h*)(Wh + (size_t)(n0 + r) * Kp + kc + q);
            *(v8h*)&Bs[r][q]     = pw[0];
            *(v8h*)&Bs[r][q + 8] = pw[1];
        }
#endif
        __syncthreads();

        // ---- fragment gather per ISA 7.12.2 + two WMMAs (A reuse) ----
        union { v16h v; v8h h[2]; } av, b0, b1;
        av.h[0] = *(const v8h*)&As[wid*16 + idx][hf*8];
        av.h[1] = *(const v8h*)&As[wid*16 + idx][16 + hf*8];
        b0.h[0] = *(const v8h*)&Bs[idx][hf*8];
        b0.h[1] = *(const v8h*)&Bs[idx][16 + hf*8];
        b1.h[0] = *(const v8h*)&Bs[16 + idx][hf*8];
        b1.h[1] = *(const v8h*)&Bs[16 + idx][16 + hf*8];
        acc0 = __builtin_amdgcn_wmma_f32_16x16x32_f16(
                   false, av.v, false, b0.v, (short)0, acc0, false, false);
        acc1 = __builtin_amdgcn_wmma_f32_16x16x32_f16(
                   false, av.v, false, b1.v, (short)0, acc1, false, false);
        __syncthreads();
    }

    // ---- epilogue: C/D layout VGPR j <-> M = j + 8*hf, N = lane&15 ----
    const int rbase = row0 + wid*16 + (hf ? 8 : 0);
    const int col0 = n0 + idx, col1 = n0 + 16 + idx;
    const float bv0 = bias ? bias[col0] : 0.f;
    const float bv1 = bias ? bias[col1] : 0.f;
#pragma unroll
    for (int j = 0; j < 8; ++j) {
        int row = rbase + j;
        float u0 = acc0[j] + bv0;
        float u1 = acc1[j] + bv1;
        if (act == 1) { u0 = u0 > 0.f ? u0 : 0.f; u1 = u1 > 0.f ? u1 : 0.f; }
        float* p0 = &O[(size_t)row * ldo + col0];
        float* p1 = &O[(size_t)row * ldo + col1];
        if (accmode == 0)      { *p0 = u0;        *p1 = u1; }
        else if (accmode == 1) { *p0 += u0;       *p1 += u1; }
        else                   { *p0 = *p0 - u0;  *p1 = *p1 - u1; }
    }
}

// ---------------------------------------------------------------------------
// Weight pack: f32 (N,K) -> f16 (Npad,Kp) zero padded.
// ---------------------------------------------------------------------------
__global__ void pack_w_f16(const float* __restrict__ src, _Float16* __restrict__ dst,
                           int N, int K, int Npad, int Kp)
{
    int i = blockIdx.x * blockDim.x + threadIdx.x;
    if (i >= Npad * Kp) return;
    int r = i / Kp, c = i % Kp;
    float v = (r < N && c < K) ? src[r * K + c] : 0.f;
    dst[i] = (_Float16)v;
}

// ---------------------------------------------------------------------------
// DFT matrices (f16), all 64x64 (rfft rows >=33 zeroed).
// ---------------------------------------------------------------------------
__global__ void gen_dft_kernel(_Float16* fhc, _Float16* fhs,
                               _Float16* ihc, _Float16* ihs,
                               _Float16* iwc, _Float16* iws,
                               _Float16* wc,  _Float16* ws_)
{
    int i = blockIdx.x * blockDim.x + threadIdx.x;
    if (i >= 4096) return;
    const float STEP = 0.09817477042468103f;  // 2*pi/64
    int r = i >> 6, c = i & 63;
    float th = (float)((r * c) & 63) * STEP;
    float cv = __cosf(th), sv = __sinf(th);
    fhc[i] = (_Float16)cv;          fhs[i] = (_Float16)(-sv);
    ihc[i] = (_Float16)(cv / 64.f); ihs[i] = (_Float16)(sv / 64.f);
    {   // inverse width: r = w', c = n (33 valid bins, hermitian fold)
        int n = c;
        float a = (n == 0 || n == 32) ? 1.f : 2.f;
        bool ok = n < 33;
        iwc[i] = (_Float16)(ok ?  a * cv / 64.f : 0.f);
        iws[i] = (_Float16)(ok ? -a * sv / 64.f : 0.f);
    }
    {   // forward width rfft: row n=r (rows >=33 zero), col w=c
        bool ok = r < 33;
        wc [i] = (_Float16)(ok ?  cv : 0.f);
        ws_[i] = (_Float16)(ok ? -sv : 0.f);
    }
}

// ---------------------------------------------------------------------------
// Small elementwise / permute kernels
// ---------------------------------------------------------------------------
__global__ void add_eps_kernel(const float* fm, const float* fl,
                               float* fme, float* fle, int n)
{
    int i = blockIdx.x * blockDim.x + threadIdx.x;
    if (i >= n) return;
    fme[i] = fm[i] + 1e-8f;
    fle[i] = fl[i] + 1e-8f;
}

// (bc,h,64) -> (bc,n<33,h)
__global__ void transpose_hn_kernel(const float* __restrict__ src, float* __restrict__ dst)
{
    int i = blockIdx.x * blockDim.x + threadIdx.x;
    if (i >= B_*C_*33*64) return;
    int h = i & 63, n = (i >> 6) % 33, bc = i / (33*64);
    dst[i] = src[((size_t)bc*64 + h)*64 + n];
}

__global__ void amp_pha_kernel(const float* zrm, const float* zim,
                               const float* zrl, const float* zil,
                               float* ampA, float* phaA)
{
    int i = blockIdx.x * blockDim.x + threadIdx.x;
    if (i >= 16896 * 192) return;
    int ch = i % 192, p = i / 192;
    int n = p % 33, khb = p / 33, kh = khb & 63, b = khb >> 6;
    int c = ch % 96;
    const float* zr = (ch < 96) ? zrm : zrl;
    const float* zi = (ch < 96) ? zim : zil;
    size_t s = ((size_t)(b*96 + c)*33 + n)*64 + kh;
    float re = zr[s], im = zi[s];
    ampA[i] = sqrtf(re*re + im*im);
    phaA[i] = atan2f(im, re);
}

__global__ void polar_kernel(const float* ampO, const float* phaO,
                             float* Rr, float* Ri)
{
    int i = blockIdx.x * blockDim.x + threadIdx.x;
    if (i >= 16896 * 96) return;
    int o = i % 96, p = i / 96;
    int n = p % 33, khb = p / 33, kh = khb & 63, b = khb >> 6;
    float a = ampO[i], ph = phaO[i];
    size_t d = ((size_t)(b*96 + o)*33 + n)*64 + kh;   // transposed for ifft-h
    Rr[d] = a * __cosf(ph) + 1e-8f;
    Ri[d] = a * __sinf(ph) + 1e-8f;
}

// (bo,n<33,h) -> (bo,h,n padded to 64)
__global__ void transpose_pad_kernel(const float* __restrict__ src, float* __restrict__ dst)
{
    int i = blockIdx.x * blockDim.x + threadIdx.x;
    if (i >= B_*C_*64*64) return;
    int n = i & 63, h = (i >> 6) & 63, bo = i >> 12;
    dst[i] = (n < 33) ? src[((size_t)bo*33 + n)*64 + h] : 0.f;
}

// |irfft2| + channel shuffle; emit fm/fl/dm/dl in (b,h,w,c) layout.
__global__ void fim_finalize_kernel(const float* __restrict__ ffraw,
                                    const float* __restrict__ fm,
                                    const float* __restrict__ fl,
                                    float* fmh, float* flh, float* dmh, float* dlh)
{
    int i = blockIdx.x * blockDim.x + threadIdx.x;
    if (i >= B_*L_*C_) return;
    int c = i % 96, l = (i / 96) % L_, b = i / (96 * L_);
    int g = c >> 1, s = c & 1;
    int co = s*48 + g;                                  // (b,2,48)->(b,48,2)
    float f  = fabsf(ffraw[((size_t)(b*96 + co))*L_ + l]);
    float vm = fm[(size_t)b*(96*L_) + (size_t)c*L_ + l]; // (b,c,h,w) reinterpret
    float vl = fl[(size_t)b*(96*L_) + (size_t)c*L_ + l];
    fmh[i] = vm; flh[i] = vl; dmh[i] = f - vm; dlh[i] = f - vl;
}

// LayerNorm over last dim, one wave per row; optional *silu(z) epilogue.
__launch_bounds__(128)
__global__ void layernorm_kernel(const float* __restrict__ x,
                                 const float* __restrict__ g,
                                 const float* __restrict__ b,
                                 float* __restrict__ o,
                                 int Cdim, float eps, int M,
                                 const float* __restrict__ z, int zld, int zoff)
{
    int wid = threadIdx.x >> 5, lane = threadIdx.x & 31;
    int row = blockIdx.x * 4 + wid;
    if (row >= M) return;
    int E = Cdim >> 5;
    float v[6];
    float s = 0.f;
    for (int e = 0; e < E; ++e) { v[e] = x[(size_t)row*Cdim + e*32 + lane]; s += v[e]; }
    for (int m = 16; m >= 1; m >>= 1) s += __shfl_xor(s, m, 32);
    float mu = s / (float)Cdim;
    float vs = 0.f;
    for (int e = 0; e < E; ++e) { float d = v[e] - mu; vs += d*d; }
    for (int m = 16; m >= 1; m >>= 1) vs += __shfl_xor(vs, m, 32);
    float inv = rsqrtf(vs / (float)Cdim + eps);
    for (int e = 0; e < E; ++e) {
        int c = e*32 + lane;
        float y = (v[e] - mu) * inv * g[c] + b[c];
        if (z) { float zz = z[(size_t)row*zld + zoff + c]; y *= zz / (1.f + __expf(-zz)); }
        o[(size_t)row*Cdim + c] = y;
    }
}

// depthwise 3x3 conv (SAME) + bias + silu, (b,h,w,384)[:,:192] -> (b,d,l)
__global__ void conv_silu_kernel(const float* __restrict__ xz,
                                 const float* __restrict__ cw,
                                 const float* __restrict__ cb,
                                 float* __restrict__ xi)
{
    int i = blockIdx.x * blockDim.x + threadIdx.x;
    if (i >= B_*DI_*L_) return;
    int l = i % L_, d = (i / L_) % DI_, b = i / (L_*DI_);
    int h = l >> 6, w = l & 63;
    float acc = 0.f;
    for (int kh = 0; kh < 3; ++kh)
        for (int kw = 0; kw < 3; ++kw) {
            int hh = h + kh - 1, ww = w + kw - 1;
            if (hh < 0 || hh > 63 || ww < 0 || ww > 63) continue;
            acc += cw[d*9 + kh*3 + kw] * xz[((size_t)(b*H_ + hh)*W_ + ww)*384 + d];
        }
    acc += cb[d];
    xi[i] = acc / (1.f + __expf(-acc));
}

// 4 scan directions, layout (k,b,l,d).
__global__ void build_xs_kernel(const float* __restrict__ xi, float* __restrict__ xs)
{
    int i = blockIdx.x * blockDim.x + threadIdx.x;
    if (i >= B_*L_*DI_) return;
    int d = i % DI_, l = (i / DI_) % L_, b = i / (DI_*L_);
    size_t src = (size_t)(b*DI_ + d) * L_;
    int lt  = ((l & 63) << 6) | (l >> 6);
    int lr  = 4095 - l;
    int ltr = ((lr & 63) << 6) | (lr >> 6);
    size_t o = ((size_t)b*L_ + l)*DI_ + d;
    size_t stride = (size_t)B_*L_*DI_;
    xs[0*stride + o] = xi[src + l];
    xs[1*stride + o] = xi[src + lt];
    xs[2*stride + o] = xi[src + lr];
    xs[3*stride + o] = xi[src + ltr];
}

// Selective scan, fused delta projection + softplus, 32 blocks x 192 threads.
// The per-step 64-float coefficient row (dts|B|C) is double-buffered through
// async global->LDS copies so step l+1 streams in while step l computes.
__launch_bounds__(192)
__global__ void scan_kernel(const float* __restrict__ xd,
                            const float* __restrict__ xs,
                            const float* __restrict__ dt_w,
                            const float* __restrict__ dt_b,
                            const float* __restrict__ a_logs,
                            const float* __restrict__ d_skip,
                            float* __restrict__ ys)
{
    int kb = blockIdx.x, k = kb >> 3;
    int d = threadIdx.x;
    __shared__ __align__(16) float srow[2][64];
    float A[NS_], hst[NS_], dtw[R_];
    for (int n = 0; n < NS_; ++n) {
        A[n] = -__expf(a_logs[((size_t)(k*DI_) + d)*NS_ + n]);
        hst[n] = 0.f;
    }
    for (int r = 0; r < R_; ++r) dtw[r] = dt_w[((size_t)(k*DI_) + d)*R_ + r];
    float dtb = dt_b[k*DI_ + d];
    float dsk = d_skip[k*DI_ + d];
    size_t base = (size_t)kb * L_;
#if ASYNC_ASM
    unsigned int lds0 = (unsigned int)(unsigned long long)(void*)&srow[0][0];
    // prologue: stage row 0 into buffer 0
    if (d < 16) {
        unsigned long long ga = (unsigned long long)(const void*)(xd + base*64 + d*4);
        unsigned int la = lds0 + (unsigned int)(d*16);
        asm volatile("global_load_async_to_lds_b128 %0, %1, off"
                     :: "v"(la), "v"(ga) : "memory");
    }
    if (d < 32) asm volatile("s_wait_asynccnt 0x0" ::: "memory");
    __syncthreads();
#endif
    for (int l = 0; l < L_; ++l) {
        int buf = l & 1;
#if ASYNC_ASM
        if (l + 1 < L_ && d < 16) {   // prefetch next row into other buffer
            unsigned long long ga =
                (unsigned long long)(const void*)(xd + (base + l + 1)*64 + d*4);
            unsigned int la = lds0 + (unsigned int)((buf ^ 1)*256 + d*16);
            asm volatile("global_load_async_to_lds_b128 %0, %1, off"
                         :: "v"(la), "v"(ga) : "memory");
        }
#else
        if (d < 64) srow[buf][d] = xd[(base + l)*64 + d];
        __syncthreads();
#endif
        const float* sr = srow[buf];
        float a = dtb;
        for (int r = 0; r < R_; ++r) a += sr[r] * dtw[r];
        float dt = (a > 20.f) ? a : log1pf(__expf(a));
        size_t rd = (base + l)*DI_ + d;
        float u = xs[rd];
        float du = dt * u, y = dsk * u;
#pragma unroll
        for (int n = 0; n < NS_; ++n) {
            hst[n] = __expf(dt * A[n]) * hst[n] + du * sr[R_ + n];
            y += hst[n] * sr[R_ + NS_ + n];
        }
        ys[rd] = y;
#if ASYNC_ASM
        if (d < 32) asm volatile("s_wait_asynccnt 0x0" ::: "memory");
#endif
        __syncthreads();
    }
}

// Merge the 4 directional scans back into (b,l,d).
__global__ void merge_y_kernel(const float* __restrict__ ys, float* __restrict__ y)
{
    int i = blockIdx.x * blockDim.x + threadIdx.x;
    if (i >= B_*L_*DI_) return;
    int d = i % DI_, l = (i / DI_) % L_, b = i / (DI_*L_);
    size_t stride = (size_t)B_*L_*DI_;
    size_t ob = (size_t)b*L_*DI_ + d;
    int lt = ((l & 63) << 6) | (l >> 6);
    float v = ys[0*stride + ob + (size_t)l*DI_]
            + ys[2*stride + ob + (size_t)(4095 - l)*DI_]
            + ys[1*stride + ob + (size_t)lt*DI_]
            + ys[3*stride + ob + (size_t)(4095 - lt)*DI_];
    y[(size_t)(b*L_ + l)*DI_ + d] = v;
}

__global__ void combine_kernel(const float* fmh, const float* flh, const float* dmh,
                               const float* ssA, const float* ssB,
                               const float* ssC, const float* ssD,
                               const float* alpha, const float* beta,
                               float* hm, float* hl, float* cat)
{
    int i = blockIdx.x * blockDim.x + threadIdx.x;
    if (i >= B_*L_*C_) return;
    int c = i % 96;
    size_t p = (size_t)i / 96;
    float fm1 = fmh[i] + ssA[i];
    float fl1 = flh[i] + ssB[i];
    float dm1 = dmh[i] + ssC[i];
    float dl1 = dmh[i] + ssD[i];           // reference uses dm + ss2d(ln4(dl))
    float vhm = fm1 + alpha[0]*dm1 + flh[i];
    float vhl = fl1 + beta[0]*dl1 + fmh[i];
    hm[i] = vhm; hl[i] = vhl;
    cat[p*192 + c] = vhm; cat[p*192 + 96 + c] = vhl;
}

__global__ void final_perm_kernel(const float* __restrict__ fp, float* __restrict__ out3)
{
    int i = blockIdx.x * blockDim.x + threadIdx.x;
    if (i >= B_*L_*C_) return;
    int c = i % 96, l = (i / 96) % L_, b = i / (96 * L_);
    out3[(size_t)b*(L_*C_) + (size_t)c*L_ + l] = fp[i];
}

// ---------------------------------------------------------------------------
// Host orchestration
// ---------------------------------------------------------------------------
static inline int cdiv(long long a, int b) { return (int)((a + b - 1) / b); }

static void gemm(hipStream_t s, const float* A, const _Float16* W, const float* bias,
                 float* O, int M, int N, int Kd, int Kp, int ldo, int act, int accmode)
{
    dim3 grid(N / 32, M / 64);
    gemm_wmma_f16<<<grid, 128, 0, s>>>(A, W, bias, O, Kd, Kp, ldo, act, accmode);
}

struct Ss2dTmp {
    float *xln, *xz, *xi, *xs, *xd, *ys, *ypre, *yln;
};

static void run_ss2d(hipStream_t s, const float* xin_hwc,
                     const float* lng, const float* lnb,
                     const _Float16* w_in, const _Float16* w_xp, const _Float16* w_out,
                     const float* conv_w, const float* conv_b,
                     const float* dt_w, const float* dt_b,
                     const float* a_logs, const float* d_skip,
                     const float* on_g, const float* on_b,
                     const Ss2dTmp& t, float* out96)
{
    const int M = B_ * L_;
    layernorm_kernel<<<M/4, 128, 0, s>>>(xin_hwc, lng, lnb, t.xln, 96, 1e-6f, M,
                                         nullptr, 0, 0);
    gemm(s, t.xln, w_in, nullptr, t.xz, M, 384, 96, 96, 384, 0, 0);
    conv_silu_kernel<<<cdiv((long long)B_*DI_*L_, 256), 256, 0, s>>>(t.xz, conv_w, conv_b, t.xi);
    build_xs_kernel<<<cdiv((long long)B_*L_*DI_, 256), 256, 0, s>>>(t.xi, t.xs);
    for (int k = 0; k < K_; ++k)
        gemm(s, t.xs + (size_t)k*M*DI_, w_xp + (size_t)k*64*192, nullptr,
             t.xd + (size_t)k*M*64, M, 64, 192, 192, 64, 0, 0);
    scan_kernel<<<K_*B_, 192, 0, s>>>(t.xd, t.xs, dt_w, dt_b, a_logs, d_skip, t.ys);
    merge_y_kernel<<<cdiv((long long)B_*L_*DI_, 256), 256, 0, s>>>(t.ys, t.ypre);
    layernorm_kernel<<<M/4, 128, 0, s>>>(t.ypre, on_g, on_b, t.yln, 192, 1e-5f, M,
                                         t.xz, 384, 192);
    gemm(s, t.yln, w_out, nullptr, out96, M, 96, 192, 192, 96, 0, 0);
}

extern "C" void kernel_launch(void* const* d_in, const int* in_sizes, int n_in,
                              void* d_out, int out_size, void* d_ws, size_t ws_size,
                              hipStream_t stream)
{
    (void)in_sizes; (void)n_in; (void)out_size; (void)ws_size;
    const float* feature_m  = (const float*)d_in[0];
    const float* feature_l  = (const float*)d_in[1];
    const float* fim_w1     = (const float*)d_in[2];
    const float* fim_b1     = (const float*)d_in[3];
    const float* fim_w2     = (const float*)d_in[4];
    const float* fim_b2     = (const float*)d_in[5];
    const float* ln_g[4]    = {(const float*)d_in[6],  (const float*)d_in[8],
                               (const float*)d_in[10], (const float*)d_in[12]};
    const float* ln_b[4]    = {(const float*)d_in[7],  (const float*)d_in[9],
                               (const float*)d_in[11], (const float*)d_in[13]};
    const float* in_proj_w  = (const float*)d_in[14];
    const float* conv_w     = (const float*)d_in[15];
    const float* conv_b     = (const float*)d_in[16];
    const float* x_proj_w   = (const float*)d_in[17];
    const float* dt_w       = (const float*)d_in[18];
    const float* dt_b       = (const float*)d_in[19];
    const float* a_logs     = (const float*)d_in[20];
    const float* d_skip     = (const float*)d_in[21];
    const float* out_norm_g = (const float*)d_in[22];
    const float* out_norm_b = (const float*)d_in[23];
    const float* out_proj_w = (const float*)d_in[24];
    const float* lin_w      = (const float*)d_in[25];
    const float* lin_b      = (const float*)d_in[26];
    const float* alpha      = (const float*)d_in[27];
    const float* beta       = (const float*)d_in[28];

    const int M  = B_ * L_;                // 32768
    const size_t PC = (size_t)M * C_;      // 3145728 elems per (b,n,c) tensor

    char* wsb = (char*)d_ws;
    size_t off = 0;
    auto alloc = [&](size_t bytes) -> void* {
        void* p = wsb + off;
        off += (bytes + 255) & ~(size_t)255;
        return p;
    };

    // ---- persistent arena ----
    _Float16* w_in  = (_Float16*)alloc((size_t)384*96*2);
    _Float16* w_xp  = (_Float16*)alloc((size_t)4*64*192*2);
    _Float16* w_out = (_Float16*)alloc((size_t)96*192*2);
    _Float16* w_lin = (_Float16*)alloc((size_t)96*192*2);
    _Float16* w_f1  = (_Float16*)alloc((size_t)96*192*2);
    _Float16* w_f2  = (_Float16*)alloc((size_t)96*192*2);
    _Float16* fhc = (_Float16*)alloc(4096*2), *fhs = (_Float16*)alloc(4096*2);
    _Float16* ihc = (_Float16*)alloc(4096*2), *ihs = (_Float16*)alloc(4096*2);
    _Float16* iwc = (_Float16*)alloc(4096*2), *iws = (_Float16*)alloc(4096*2);
    _Float16* dwc = (_Float16*)alloc(4096*2), *dws = (_Float16*)alloc(4096*2);
    float* fmh = (float*)alloc(PC*4); float* flh = (float*)alloc(PC*4);
    float* dmh = (float*)alloc(PC*4); float* dlh = (float*)alloc(PC*4);
    float* ss[4]; for (int i = 0; i < 4; ++i) ss[i] = (float*)alloc(PC*4);
    float* cat  = (float*)alloc((size_t)M*192*4);
    float* fpre = (float*)alloc(PC*4);
    size_t mark = off;

    // ---- weight packing + DFT generation ----
    pack_w_f16<<<cdiv(384*96,256),256,0,stream>>>(in_proj_w, w_in, 384, 96, 384, 96);
    for (int k = 0; k < 4; ++k)
        pack_w_f16<<<cdiv(64*192,256),256,0,stream>>>(x_proj_w + (size_t)k*38*192,
                                                      w_xp + (size_t)k*64*192, 38, 192, 64, 192);
    pack_w_f16<<<cdiv(96*192,256),256,0,stream>>>(out_proj_w, w_out, 96, 192, 96, 192);
    pack_w_f16<<<cdiv(96*192,256),256,0,stream>>>(lin_w,  w_lin, 96, 192, 96, 192);
    pack_w_f16<<<cdiv(96*192,256),256,0,stream>>>(fim_w1, w_f1,  96, 192, 96, 192);
    pack_w_f16<<<cdiv(96*192,256),256,0,stream>>>(fim_w2, w_f2,  96, 192, 96, 192);
    gen_dft_kernel<<<16,256,0,stream>>>(fhc, fhs, ihc, ihs, iwc, iws, dwc, dws);

    // ================= FIM (FFT fusion) phase =================
    {
        const int M1 = B_*C_*64;   // 49152 rows of width-64
        const int M2 = B_*C_*33;   // 25344 rows of height-64
        float* fme  = (float*)alloc(PC*4);
        float* fle  = (float*)alloc(PC*4);
        float* y1rm = (float*)alloc((size_t)M1*64*4), *y1im = (float*)alloc((size_t)M1*64*4);
        float* y1rl = (float*)alloc((size_t)M1*64*4), *y1il = (float*)alloc((size_t)M1*64*4);
        float* trm  = (float*)alloc((size_t)M2*64*4), *tim  = (float*)alloc((size_t)M2*64*4);
        float* trl  = (float*)alloc((size_t)M2*64*4), *til  = (float*)alloc((size_t)M2*64*4);
        float* zrm  = (float*)alloc((size_t)M2*64*4), *zim  = (float*)alloc((size_t)M2*64*4);
        float* zrl  = (float*)alloc((size_t)M2*64*4), *zil  = (float*)alloc((size_t)M2*64*4);
        float* ampA = (float*)alloc((size_t)16896*192*4), *phaA = (float*)alloc((size_t)16896*192*4);
        float* ampO = (float*)alloc((size_t)16896*96*4),  *phaO = (float*)alloc((size_t)16896*96*4);
        float* Rr   = (float*)alloc((size_t)M2*64*4), *Ri = (float*)alloc((size_t)M2*64*4);
        float* z2r  = (float*)alloc((size_t)M2*64*4), *z2i = (float*)alloc((size_t)M2*64*4);
        float* a3r  = (float*)alloc((size_t)M1*64*4), *a3i = (float*)alloc((size_t)M1*64*4);
        float* ffr  = (float*)alloc((size_t)M1*64*4);

        add_eps_kernel<<<cdiv(PC,256),256,0,stream>>>(feature_m, feature_l, fme, fle, (int)PC);
        // width rfft as GEMM (rows >=33 of W are zero)
        gemm(stream, fme, dwc, nullptr, y1rm, M1, 64, 64, 64, 64, 0, 0);
        gemm(stream, fme, dws, nullptr, y1im, M1, 64, 64, 64, 64, 0, 0);
        gemm(stream, fle, dwc, nullptr, y1rl, M1, 64, 64, 64, 64, 0, 0);
        gemm(stream, fle, dws, nullptr, y1il, M1, 64, 64, 64, 64, 0, 0);
        int tn = B_*C_*33*64;
        transpose_hn_kernel<<<cdiv(tn,256),256,0,stream>>>(y1rm, trm);
        transpose_hn_kernel<<<cdiv(tn,256),256,0,stream>>>(y1im, tim);
        transpose_hn_kernel<<<cdiv(tn,256),256,0,stream>>>(y1rl, trl);
        transpose_hn_kernel<<<cdiv(tn,256),256,0,stream>>>(y1il, til);
        // height fft (complex)
        gemm(stream, trm, fhc, nullptr, zrm, M2, 64, 64, 64, 64, 0, 0);
        gemm(stream, tim, fhs, nullptr, zrm, M2, 64, 64, 64, 64, 0, -1);
        gemm(stream, trm, fhs, nullptr, zim, M2, 64, 64, 64, 64, 0, 0);
        gemm(stream, tim, fhc, nullptr, zim, M2, 64, 64, 64, 64, 0, 1);
        gemm(stream, trl, fhc, nullptr, zrl, M2, 64, 64, 64, 64, 0, 0);
        gemm(stream, til, fhs, nullptr, zrl, M2, 64, 64, 64, 64, 0, -1);
        gemm(stream, trl, fhs, nullptr, zil, M2, 64, 64, 64, 64, 0, 0);
        gemm(stream, til, fhc, nullptr, zil, M2, 64, 64, 64, 64, 0, 1);
        // amplitude / phase mixing MLP (relu)
        amp_pha_kernel<<<cdiv(16896*192,256),256,0,stream>>>(zrm, zim, zrl, zil, ampA, phaA);
        gemm(stream, ampA, w_f1, fim_b1, ampO, 16896, 96, 192, 192, 96, 1, 0);
        gemm(stream, phaA, w_f2, fim_b2, phaO, 16896, 96, 192, 192, 96, 1, 0);
        polar_kernel<<<cdiv(16896*96,256),256,0,stream>>>(ampO, phaO, Rr, Ri);
        // inverse height fft (complex)
        gemm(stream, Rr, ihc, nullptr, z2r, M2, 64, 64, 64, 64, 0, 0);
        gemm(stream, Ri, ihs, nullptr, z2r, M2, 64, 64, 64, 64, 0, -1);
        gemm(stream, Rr, ihs, nullptr, z2i, M2, 64, 64, 64, 64, 0, 0);
        gemm(stream, Ri, ihc, nullptr, z2i, M2, 64, 64, 64, 64, 0, 1);
        int pn = B_*C_*64*64;
        transpose_pad_kernel<<<cdiv(pn,256),256,0,stream>>>(z2r, a3r);
        transpose_pad_kernel<<<cdiv(pn,256),256,0,stream>>>(z2i, a3i);
        // inverse width rfft (real output)
        gemm(stream, a3r, iwc, nullptr, ffr, M1, 64, 64, 64, 64, 0, 0);
        gemm(stream, a3i, iws, nullptr, ffr, M1, 64, 64, 64, 64, 0, 1);
        fim_finalize_kernel<<<cdiv(PC,256),256,0,stream>>>(ffr, feature_m, feature_l,
                                                           fmh, flh, dmh, dlh);
    }

    // ================= SS2D phase (reuse FFT scratch) =================
    off = mark;
    Ss2dTmp t;
    t.xln  = (float*)alloc(PC*4);
    t.xz   = (float*)alloc((size_t)M*384*4);
    t.xi   = (float*)alloc((size_t)B_*DI_*L_*4);
    t.xs   = (float*)alloc((size_t)K_*M*DI_*4);
    t.xd   = (float*)alloc((size_t)K_*M*64*4);
    t.ys   = (float*)alloc((size_t)K_*M*DI_*4);
    t.ypre = (float*)alloc((size_t)M*DI_*4);
    t.yln  = (float*)alloc((size_t)M*DI_*4);

    const float* xin[4] = {fmh, flh, dmh, dlh};
    for (int i = 0; i < 4; ++i)
        run_ss2d(stream, xin[i], ln_g[i], ln_b[i], w_in, w_xp, w_out,
                 conv_w, conv_b, dt_w, dt_b, a_logs, d_skip,
                 out_norm_g, out_norm_b, t, ss[i]);

    // ================= final assembly =================
    float* out = (float*)d_out;
    float* out_hm = out;
    float* out_hl = out + PC;
    float* out_fi = out + 2*PC;
    combine_kernel<<<cdiv(PC,256),256,0,stream>>>(fmh, flh, dmh,
                                                  ss[0], ss[1], ss[2], ss[3],
                                                  alpha, beta, out_hm, out_hl, cat);
    gemm(stream, cat, w_lin, lin_b, fpre, M, 96, 192, 192, 96, 0, 0);
    final_perm_kernel<<<cdiv(PC,256),256,0,stream>>>(fpre, out_fi);
}
